// AttentionSelf_52853867545194
// MI455X (gfx1250) — matrix operand order repaired and checked by reading, not compiled, not run on hardware
//
#include <hip/hip_runtime.h>
#include <hip/hip_bf16.h>

// ---------------------------------------------------------------------------
// Self-attention, B=4 S=2048 D=1024, via V_WMMA_F32_16X16X32_F16 (gfx1250),
// with TENSOR_LOAD_TO_LDS (TDM) staging of the Q tile in the fused kernel.
// Workspace layout (assumes ws_size >= ~74 MB):
//   x16   : [8192][1024] f16          16.78 MB
//   WqT/WkT/WvT : [1024][1024] f16     3 x 2 MB   (transposed: row n holds k)
//   Q16,K16 : [B*S][1024] f16          2 x 16.78 MB
//   Vt16  : [B][1024][2048] f16        16.78 MB   (transposed V)
// ---------------------------------------------------------------------------

typedef _Float16 half_t;
typedef __attribute__((ext_vector_type(16))) _Float16 v16h;
typedef __attribute__((ext_vector_type(8)))  _Float16 v8h;
typedef __attribute__((ext_vector_type(4)))  _Float16 v4h;
typedef __attribute__((ext_vector_type(8)))  float    v8f;
typedef __attribute__((ext_vector_type(4)))  unsigned int u32x4;
typedef __attribute__((ext_vector_type(8)))  int      i32x8;
typedef __attribute__((ext_vector_type(4)))  int      i32x4;

#if defined(__has_builtin)
#  if __has_builtin(__builtin_amdgcn_tensor_load_to_lds)
#    define USE_TDM 1
#  endif
#endif
#ifndef USE_TDM
#  define USE_TDM 0
#endif

union AFrag { v16h v; v8h h[2]; };

// A-matrix fragment (16x32 f16) from row-major src, rows at `base`, ld elems.
// ISA layout: lanes0-15 row=lane, k = {0..7, 16..23}; lanes16-31 k = {8..15, 24..31}.
__device__ __forceinline__ v16h load_a_frag(const half_t* base, int ld) {
  int lane = threadIdx.x & 31;
  int r = lane & 15, sel = lane >> 4;
  const half_t* p = base + (size_t)r * ld + sel * 8;
  AFrag f;
  f.h[0] = *(const v8h*)p;
  f.h[1] = *(const v8h*)(p + 16);
  return f.v;
}

// B-matrix fragment (32x16 f16), B[k][n] = T[n][k] with T row-major (ld elems).
// ISA layout: lanes0-15 col=lane, k = {0..7, 16..23}; lanes16-31 col=lane-16, k = {8..15, 24..31}.
__device__ __forceinline__ v16h load_b_frag(const half_t* base, int ld) {
  int lane = threadIdx.x & 31;
  int n = lane & 15, sel = lane >> 4;
  const half_t* p = base + (size_t)n * ld + sel * 8;
  AFrag f;
  f.h[0] = *(const v8h*)p;
  f.h[1] = *(const v8h*)(p + 16);
  return f.v;
}

__device__ __forceinline__ v8f wmma_f16(v16h a, v16h b, v8f c) {
  return __builtin_amdgcn_wmma_f32_16x16x32_f16(false, a, false, b,
                                                (short)0, c, false, false);
}

// ---------------------------------------------------------------------------
__global__ void __launch_bounds__(256)
k_f32_to_f16(const float* __restrict__ in, half_t* __restrict__ out, int n4) {
  int stride = gridDim.x * blockDim.x;
  for (int i = blockIdx.x * blockDim.x + threadIdx.x; i < n4; i += stride) {
    float4 v = ((const float4*)in)[i];
    v4h h;
    h[0] = (half_t)v.x; h[1] = (half_t)v.y; h[2] = (half_t)v.z; h[3] = (half_t)v.w;
    ((v4h*)out)[i] = h;
  }
}

// W [1024][1024] f32 row-major (k,n) -> Wt [n][k] f16
__global__ void __launch_bounds__(256)
k_w_transpose(const float* __restrict__ W, half_t* __restrict__ Wt) {
  __shared__ float tile[32][33];
  int n0 = blockIdx.x * 32, k0 = blockIdx.y * 32;
  int tx = threadIdx.x & 31, ty = threadIdx.x >> 5;   // 32 x 8
  for (int r = ty; r < 32; r += 8)
    tile[r][tx] = W[(size_t)(k0 + r) * 1024 + n0 + tx];
  __syncthreads();
  for (int r = ty; r < 32; r += 8)
    Wt[(size_t)(n0 + r) * 1024 + k0 + tx] = (half_t)tile[tx][r];
}

// ---------------------------------------------------------------------------
// C[M=8192, N=1024] = X[8192,1024] @ W[1024,1024] + bias, f16 out.
// 8 waves, wave tile 32x64 (2x4 frags), WG tile 128x128.
// transpose_out: write Vt[b][n][s] instead of OUT[m][n].
__global__ void __launch_bounds__(256)
k_qkv_gemm(const half_t* __restrict__ X, const half_t* __restrict__ Wt,
           const float* __restrict__ bias, half_t* __restrict__ OUT,
           int transpose_out) {
  const int ld = 1024;
  int wave = threadIdx.x >> 5, lane = threadIdx.x & 31;
  int wr = wave >> 1, wc = wave & 1;
  int m0 = blockIdx.x * 128 + wr * 32;
  int n0 = blockIdx.y * 128 + wc * 64;

  v8f acc[2][4] = {};
  for (int kc = 0; kc < 1024; kc += 32) {
    v16h a0 = load_a_frag(X + (size_t)m0 * ld + kc, ld);
    v16h a1 = load_a_frag(X + (size_t)(m0 + 16) * ld + kc, ld);
#pragma unroll
    for (int f = 0; f < 4; ++f) {
      v16h b = load_b_frag(Wt + (size_t)(n0 + f * 16) * ld + kc, ld);
      acc[0][f] = wmma_f16(a0, b, acc[0][f]);
      acc[1][f] = wmma_f16(a1, b, acc[1][f]);
    }
  }

  int col = lane & 15, sel = lane >> 4;
#pragma unroll
  for (int mf = 0; mf < 2; ++mf) {
    int mbase = m0 + mf * 16 + sel * 8;
#pragma unroll
    for (int f = 0; f < 4; ++f) {
      int n = n0 + f * 16 + col;
      float bv = bias[n];
      if (!transpose_out) {
#pragma unroll
        for (int j = 0; j < 8; ++j)
          OUT[(size_t)(mbase + j) * 1024 + n] = (half_t)(acc[mf][f][j] + bv);
      } else {
        int b = mbase >> 11, s = mbase & 2047;   // m = b*2048 + s
        v8h pack;
#pragma unroll
        for (int j = 0; j < 8; ++j) pack[j] = (half_t)(acc[mf][f][j] + bv);
        *(v8h*)(OUT + ((size_t)b * 1024 + n) * 2048 + s) = pack;  // contiguous in s
      }
    }
  }
}

// ---------------------------------------------------------------------------
// Fused scores + softmax + P@V per 16-query tile. 8 waves.
// LDS: sQ 16x1024 f16 (32KB) | sS 16x2048 f32 (128KB) | sP 16x2048 f16 (64KB)
__global__ void __launch_bounds__(256)
k_attn(const half_t* __restrict__ Q, const half_t* __restrict__ K,
       const half_t* __restrict__ Vt, float* __restrict__ OUT) {
  extern __shared__ char smem[];
  half_t* sQ = (half_t*)smem;                        // 16 x 1024
  float*  sS = (float*)(smem + 32768);               // 16 x 2048
  half_t* sP = (half_t*)(smem + 32768 + 131072);     // 16 x 2048

  int tid = threadIdx.x;
  int wave = tid >> 5, lane = tid & 31;
  int b  = blockIdx.x >> 7;            // 128 query tiles per batch
  int q0 = (blockIdx.x & 127) * 16;

  const half_t* Qb = Q  + ((size_t)b * 2048 + q0) * 1024;
  const half_t* Kb = K  + (size_t)b * 2048 * 1024;
  const half_t* Vb = Vt + (size_t)b * 1024 * 2048;

  // ---- Stage Q tile (one contiguous 32 KB block) into LDS ----
#if USE_TDM
  if (wave == 0) {
    // 1-D TDM descriptor: 4096 elements x 8 bytes, tile == tensor.
    unsigned long long ga = (unsigned long long)(size_t)Qb;
    unsigned lds_off = (unsigned)(size_t)(void*)sQ;   // addr[31:0] == LDS offset
    u32x4 g0;
    g0[0] = 1u;                                       // count=1 (valid user D#)
    g0[1] = lds_off;                                  // lds_addr (bytes)
    g0[2] = (unsigned)ga;                             // global_addr[31:0]
    g0[3] = (unsigned)((ga >> 32) & 0x01FFFFFFu) | (2u << 30);  // addr[56:32]|type=2
    i32x8 g1 = (i32x8)0;
    g1[0] = (3 << 16);                                // data_size = 8 bytes
    g1[1] = (int)(4096u << 16);                       // tensor_dim0 = 4096 (lo16)
    g1[2] = (int)(1u << 16);                          // dim0 hi=0 | tensor_dim1 = 1
    g1[3] = (int)(4096u << 16);                       // dim1 hi=0 | tile_dim0 = 4096
    g1[5] = 4096;                                     // tensor_dim0_stride
    i32x4 gz = (i32x4)0;
#  if __has_include(<hip/amd_detail/amd_gfx1250_TDM.h>)
    __builtin_amdgcn_tensor_load_to_lds(g0, g1, gz, gz, (i32x8)0, 0);
#  else
    __builtin_amdgcn_tensor_load_to_lds(g0, g1, gz, gz, 0);
#  endif
#  if __has_builtin(__builtin_amdgcn_s_wait_tensorcnt)
    __builtin_amdgcn_s_wait_tensorcnt(0);
#  else
    asm volatile("s_wait_tensorcnt 0x0" ::: "memory");
#  endif
  }
  __syncthreads();
#else
  for (int i = tid; i < 2048; i += 256)
    *(v8h*)(sQ + (size_t)i * 8) = *(const v8h*)(Qb + (size_t)i * 8);
  __syncthreads();
#endif

  // ---- Scores: wave owns key columns [wave*256, wave*256+256). ----
  // kc-outer / nf-inner: the Q A-fragment is loaded once per k-chunk and
  // reused by 16 WMMAs (16 live f32 accumulators = 128 VGPRs).
  int col = lane & 15, sel = lane >> 4;
  const half_t* Kw = Kb + (size_t)(wave * 256) * 1024;
  v8f sacc[16] = {};
  for (int kc = 0; kc < 1024; kc += 32) {
    v16h a = load_a_frag(sQ + kc, 1024);
    if (kc + 32 < 1024)
      __builtin_prefetch(Kw + kc + 32, 0, 3);         // global_prefetch_b8
#pragma unroll
    for (int nf = 0; nf < 16; ++nf) {
      v16h bk = load_b_frag(Kw + (size_t)(nf * 16) * 1024 + kc, 1024);
      sacc[nf] = wmma_f16(a, bk, sacc[nf]);
    }
  }
#pragma unroll
  for (int nf = 0; nf < 16; ++nf) {
    int n0 = wave * 256 + nf * 16;
#pragma unroll
    for (int j = 0; j < 8; ++j)
      sS[(size_t)(j + sel * 8) * 2048 + n0 + col] = sacc[nf][j];
  }
  __syncthreads();

  // ---- Softmax (reference order: softmax(raw scores) then * 1/sqrt(1024)) ----
  for (int rr = 0; rr < 2; ++rr) {
    int row = wave * 2 + rr;
    float* srow = sS + (size_t)row * 2048;
    half_t* prow = sP + (size_t)row * 2048;
    float m = -3.4e38f;
    for (int i = lane; i < 2048; i += 32) m = fmaxf(m, srow[i]);
#pragma unroll
    for (int off = 16; off; off >>= 1) m = fmaxf(m, __shfl_xor(m, off, 32));
    float sum = 0.f;
    for (int i = lane; i < 2048; i += 32) {
      float e = __expf(srow[i] - m);
      sum += e;
      prow[i] = (half_t)e;
    }
#pragma unroll
    for (int off = 16; off; off >>= 1) sum += __shfl_xor(sum, off, 32);
    float scale = 0.03125f / sum;               // 1/sqrt(1024) / rowsum
    for (int i = lane; i < 2048; i += 32)
      prow[i] = (half_t)((float)prow[i] * scale);
  }
  __syncthreads();

  // ---- Output: wave owns 128 columns -> 8 fragments, contract over 2048 ----
  const half_t* Vw = Vb + (size_t)(wave * 128) * 2048;
  v8f oacc[8] = {};
  for (int kc = 0; kc < 2048; kc += 32) {
    v16h a = load_a_frag(sP + kc, 2048);
    if (kc + 32 < 2048)
      __builtin_prefetch(Vw + kc + 32, 0, 3);
#pragma unroll
    for (int f = 0; f < 8; ++f) {
      v16h bv = load_b_frag(Vw + (size_t)(f * 16) * 2048 + kc, 2048);
      oacc[f] = wmma_f16(a, bv, oacc[f]);
    }
  }
  float* Ob = OUT + ((size_t)b * 2048 + q0) * 1024;
#pragma unroll
  for (int f = 0; f < 8; ++f) {
    int n = wave * 128 + f * 16 + col;
#pragma unroll
    for (int j = 0; j < 8; ++j)
      Ob[(size_t)(j + sel * 8) * 1024 + n] = oacc[f][j];
  }
}

// ---------------------------------------------------------------------------
extern "C" void kernel_launch(void* const* d_in, const int* in_sizes, int n_in,
                              void* d_out, int out_size, void* d_ws, size_t ws_size,
                              hipStream_t stream) {
  const float* x  = (const float*)d_in[0];
  const float* Wq = (const float*)d_in[1];
  const float* bq = (const float*)d_in[2];
  const float* Wk = (const float*)d_in[3];
  const float* bk = (const float*)d_in[4];
  const float* Wv = (const float*)d_in[5];
  const float* bv = (const float*)d_in[6];
  float* out = (float*)d_out;

  char* ws = (char*)d_ws;
  const size_t SZ_X16 = (size_t)8192 * 1024 * 2;   // 16.78 MB
  const size_t SZ_W16 = (size_t)1024 * 1024 * 2;   //  2.10 MB
  size_t off = 0;
  half_t* x16  = (half_t*)(ws + off); off += SZ_X16;
  half_t* wqt  = (half_t*)(ws + off); off += SZ_W16;
  half_t* wkt  = (half_t*)(ws + off); off += SZ_W16;
  half_t* wvt  = (half_t*)(ws + off); off += SZ_W16;
  half_t* Q16  = (half_t*)(ws + off); off += SZ_X16;
  half_t* K16  = (half_t*)(ws + off); off += SZ_X16;
  half_t* Vt16 = (half_t*)(ws + off); off += SZ_X16;  // ~73.4 MB total

  // 1) precision conversion / weight transpose
  k_f32_to_f16<<<2048, 256, 0, stream>>>(x, x16, 8192 * 1024 / 4);
  k_w_transpose<<<dim3(32, 32), 256, 0, stream>>>(Wq, wqt);
  k_w_transpose<<<dim3(32, 32), 256, 0, stream>>>(Wk, wkt);
  k_w_transpose<<<dim3(32, 32), 256, 0, stream>>>(Wv, wvt);

  // 2) QKV projections (WMMA GEMMs); V stored transposed for the P@V stage
  dim3 ggrid(64, 8);
  k_qkv_gemm<<<ggrid, 256, 0, stream>>>(x16, wqt, bq, Q16, 0);
  k_qkv_gemm<<<ggrid, 256, 0, stream>>>(x16, wkt, bk, K16, 0);
  k_qkv_gemm<<<ggrid, 256, 0, stream>>>(x16, wvt, bv, Vt16, 1);

  // 3) fused scores + softmax + P@V  (224 KB LDS / workgroup; CDNA5 allows 320 KB)
  const int attn_lds = 32768 + 131072 + 65536;
  hipFuncSetAttribute((const void*)k_attn,
                      hipFuncAttributeMaxDynamicSharedMemorySize, attn_lds);
  k_attn<<<512, 256, attn_lds, stream>>>(Q16, K16, Vt16, out);
}
